// Decoder_40759239639197
// MI455X (gfx1250) — compile-verified, matrix operand order checked by reading, not run on hardware
//
#include <hip/hip_runtime.h>
#include <stdint.h>

// ---------------------------------------------------------------------------
// Tacotron2-style decoder for MI455X (gfx1250, wave32, WMMA).
// Strategy:
//  * All big matmuls in bf16 via v_wmma_f32_16x16x32_bf16 (M=16 == batch).
//  * Weights converted to bf16 once per call; resident in 192MB L2.
//  * Sequential 200-step loop driven by stream ordering.
//  * Postnet = implicit GEMM over zero-padded time-major bf16 activations.
// ---------------------------------------------------------------------------

typedef __attribute__((ext_vector_type(16))) __bf16 v16bf;
typedef __attribute__((ext_vector_type(8)))  float  v8f;

#define B_   16
#define S_   256
#define T_   200
#define MEL_ 80
#define ENC_ 512
#define DEC_ 1024
#define ATTN_ 128
#define PRE_ 256
#define TP_  224          // padded time rows for postnet buffers
#define CIN1P_ 96         // MEL padded to multiple of 32 for WMMA K
#define BN_INV_ 0.9999950000374997f

__device__ __forceinline__ __bf16 f2bf(float f) { return (__bf16)f; }
__device__ __forceinline__ float sigf(float x) { return 1.0f / (1.0f + expf(-x)); }

// Load one 16x32 bf16 fragment (A or B operand) from a row-major matrix.
// rowptr points at this lane's matrix row (row = lane&15 within the tile).
// ISA layout: lanes 0-15 hold K {0..7,16..23}, lanes 16-31 hold K {8..15,24..31};
// each VGPR packs 2 consecutive K values -> dword index (r&3) + half*4 + (r>>2)*8.
__device__ __forceinline__ v16bf frag_load(const __bf16* __restrict__ rowptr,
                                           int k0, int half) {
  union { v16bf v; uint32_t u[8]; } f;
  const uint32_t* p = (const uint32_t*)(rowptr + k0);
  int hb = half * 4;
#pragma unroll
  for (int r = 0; r < 8; ++r) {
    int dw = (r & 3) + hb + ((r >> 2) << 3);
    f.u[r] = p[dw];
  }
  return f.v;
}

// ---------------------------------------------------------------------------
// Utility kernels
// ---------------------------------------------------------------------------
__global__ void k_zero(uint32_t* p, int n) {
  int i = blockIdx.x * blockDim.x + threadIdx.x;
  if (i < n) p[i] = 0u;
}

__global__ void k_f2bf(__bf16* dst, const float* __restrict__ src, int n) {
  int i = blockIdx.x * blockDim.x + threadIdx.x;
  if (i < n) dst[i] = f2bf(src[i]);
}

// dst[n][0:Ka] = A[n][:], dst[n][Ka:Ka+Kb] = Bm[n][:]   (bf16 concat of weights)
__global__ void k_concat2(__bf16* dst, const float* __restrict__ A, int Ka,
                          const float* __restrict__ Bm, int Kb, int N) {
  int idx = blockIdx.x * blockDim.x + threadIdx.x;
  int Kt = Ka + Kb;
  if (idx >= N * Kt) return;
  int n = idx / Kt;
  int k = idx - n * Kt;
  float v = (k < Ka) ? A[n * Ka + k] : Bm[n * Kb + (k - Ka)];
  dst[idx] = f2bf(v);
}

// conv weight (CO,CI,5) -> bf16 (CO,5,CIP), zero padded ci in [CI,CIP)
__global__ void k_convw(__bf16* dst, const float* __restrict__ src,
                        int CO, int CI, int CIP) {
  int idx = blockIdx.x * blockDim.x + threadIdx.x;
  if (idx >= CO * 5 * CIP) return;
  int co = idx / (5 * CIP);
  int rem = idx - co * 5 * CIP;
  int k = rem / CIP;
  int ci = rem - k * CIP;
  dst[idx] = (ci < CI) ? f2bf(src[(co * CI + ci) * 5 + k]) : f2bf(0.0f);
}

// mel_outputs (B,MEL,T) f32 -> xT (B,TP,CIN1P) bf16 at row t+2
__global__ void k_mel_to_xT(const float* __restrict__ mel, __bf16* xT) {
  int idx = blockIdx.x * blockDim.x + threadIdx.x;
  if (idx >= B_ * T_ * MEL_) return;
  int b = idx / (T_ * MEL_);
  int rem = idx - b * T_ * MEL_;
  int t = rem / MEL_;
  int c = rem - t * MEL_;
  xT[((size_t)(b * TP_ + t + 2)) * CIN1P_ + c] =
      f2bf(mel[(size_t)(b * MEL_ + c) * T_ + t]);
}

// ---------------------------------------------------------------------------
// keys = encoder_outputs @ Wk^T : M=4096 (b,s), N=128, K=512 (bf16 WMMA)
// grid.x = 256 m-tiles; 8 waves/block = 8 n-tiles (exactly ATTN/16)
// ---------------------------------------------------------------------------
__global__ void __launch_bounds__(256)
k_keys(const __bf16* __restrict__ Ebf, const __bf16* __restrict__ Wkbf,
       float* __restrict__ keys) {
  int lane = threadIdx.x & 31, wave = threadIdx.x >> 5;
  int m0 = blockIdx.x * 16;
  int n0 = wave * 16;
  int half = lane >> 4, l15 = lane & 15;
  const __bf16* arow = Ebf + (size_t)(m0 + l15) * ENC_;
  const __bf16* wrow = Wkbf + (size_t)(n0 + l15) * ENC_;
  v8f acc = {};
  for (int k0 = 0; k0 < ENC_; k0 += 32) {
    v16bf a = frag_load(arow, k0, half);
    v16bf b = frag_load(wrow, k0, half);
    acc = __builtin_amdgcn_wmma_f32_16x16x32_bf16(false, a, false, b,
                                                  (short)0, acc, false, false);
  }
#pragma unroll
  for (int r = 0; r < 8; ++r)
    keys[(size_t)(m0 + r + 8 * half) * ATTN_ + n0 + l15] = acc[r];
}

// ---------------------------------------------------------------------------
// Generic M=16 GEMM: Z(16,N) = A(16,K) @ W(N,K)^T   (bf16 in, f32 out)
// Used for both LSTM gate matmuls. 8 waves/block -> 8 n-tiles.
// ---------------------------------------------------------------------------
__global__ void __launch_bounds__(256)
k_gemm16(const __bf16* __restrict__ A, const __bf16* __restrict__ W,
         float* __restrict__ Z, int K, int N) {
  int lane = threadIdx.x & 31, wave = threadIdx.x >> 5;
  int n0 = (blockIdx.x * 8 + wave) * 16;
  if (n0 >= N) return;                       // wave-uniform exit
  int half = lane >> 4, l15 = lane & 15;
  const __bf16* arow = A + (size_t)l15 * K;
  const __bf16* wrow = W + (size_t)(n0 + l15) * K;
  v8f acc = {};
  for (int k0 = 0; k0 < K; k0 += 32) {
    __builtin_prefetch(wrow + k0 + 512, 0, 1);   // global_prefetch_b8: next W chunk
    v16bf a = frag_load(arow, k0, half);
    v16bf b = frag_load(wrow, k0, half);
    acc = __builtin_amdgcn_wmma_f32_16x16x32_bf16(false, a, false, b,
                                                  (short)0, acc, false, false);
  }
  int n = n0 + l15;
#pragma unroll
  for (int r = 0; r < 8; ++r)
    Z[(size_t)(r + 8 * half) * N + n] = acc[r];
}

// ---------------------------------------------------------------------------
// mel = [h1 | context](16,1536) @ mw^T(1536,80) + mb, written to (B,MEL,T) at t
// ---------------------------------------------------------------------------
__global__ void __launch_bounds__(256)
k_mel(int t, const __bf16* __restrict__ Amel, const __bf16* __restrict__ mwbf,
      const float* __restrict__ mb, float* __restrict__ out_mel) {
  int lane = threadIdx.x & 31, wave = threadIdx.x >> 5;
  int n0 = wave * 16;
  if (n0 >= MEL_) return;
  int half = lane >> 4, l15 = lane & 15;
  const __bf16* arow = Amel + (size_t)l15 * (DEC_ + ENC_);
  const __bf16* wrow = mwbf + (size_t)(n0 + l15) * (DEC_ + ENC_);
  v8f acc = {};
  for (int k0 = 0; k0 < DEC_ + ENC_; k0 += 32) {
    v16bf a = frag_load(arow, k0, half);
    v16bf b = frag_load(wrow, k0, half);
    acc = __builtin_amdgcn_wmma_f32_16x16x32_bf16(false, a, false, b,
                                                  (short)0, acc, false, false);
  }
  int n = n0 + l15;
#pragma unroll
  for (int r = 0; r < 8; ++r) {
    int bb = r + 8 * half;
    out_mel[(size_t)(bb * MEL_ + n) * T_ + t] = acc[r] + mb[n];
  }
}

// ---------------------------------------------------------------------------
// LSTM pointwise: gates -> (c,h); writes h as bf16 into next GEMM A buffers.
// 16*1024 threads.
// ---------------------------------------------------------------------------
__global__ void __launch_bounds__(256)
k_lstm_pw(const float* __restrict__ z, const float* __restrict__ bih,
          const float* __restrict__ bhh, float* __restrict__ c,
          float* __restrict__ hf32,
          __bf16* __restrict__ dstA, int strideA, int offA,
          __bf16* __restrict__ dstB, int strideB, int offB) {
  int idx = blockIdx.x * blockDim.x + threadIdx.x;
  if (idx >= B_ * DEC_) return;
  int b = idx >> 10, j = idx & 1023;
  const float* zb = z + (size_t)b * 4 * DEC_;
  float gi = zb[j]            + bih[j]            + bhh[j];
  float gf = zb[DEC_ + j]     + bih[DEC_ + j]     + bhh[DEC_ + j];
  float gg = zb[2 * DEC_ + j] + bih[2 * DEC_ + j] + bhh[2 * DEC_ + j];
  float go = zb[3 * DEC_ + j] + bih[3 * DEC_ + j] + bhh[3 * DEC_ + j];
  float cn = sigf(gf) * c[idx] + sigf(gi) * tanhf(gg);
  float hn = sigf(go) * tanhf(cn);
  c[idx] = cn;
  if (hf32) hf32[idx] = hn;
  __bf16 hb = f2bf(hn);
  dstA[(size_t)b * strideA + offA + j] = hb;
  dstB[(size_t)b * strideB + offB + j] = hb;
}

// ---------------------------------------------------------------------------
// Fused prenet + location-sensitive attention + context. 1 block per batch b.
// Writes: attn (d_out + prev_attn state), prenet/context bf16 into A buffers.
// ---------------------------------------------------------------------------
__global__ void __launch_bounds__(256)
k_attention(int t,
            const float* __restrict__ melT, const float* __restrict__ enc,
            const float* __restrict__ pw1, const float* __restrict__ pb1,
            const float* __restrict__ pw2, const float* __restrict__ pb2,
            const float* __restrict__ Wq, const float* __restrict__ keys,
            const float* __restrict__ loc_w, const float* __restrict__ loc_b,
            const float* __restrict__ Wl, const float* __restrict__ vvec,
            const float* __restrict__ h1f, float* __restrict__ prev_attn,
            float* __restrict__ out_attn,
            __bf16* __restrict__ Acat0, __bf16* __restrict__ Amel) {
  __shared__ float s_dec[MEL_];
  __shared__ float s_pre1[PRE_];
  __shared__ float s_q[ATTN_];
  __shared__ float s_pad[S_ + 30];
  __shared__ float s_attn[S_];
  __shared__ float s_red[S_];
  int tid = threadIdx.x;
  int b = blockIdx.x;

  if (tid < MEL_)
    s_dec[tid] = (t == 0) ? 0.0f : melT[(size_t)(b * MEL_ + tid) * T_ + (t - 1)];
  {
    int idx = tid - 15;
    s_pad[tid] = (idx >= 0 && idx < S_) ? prev_attn[b * S_ + idx] : 0.0f;
    if (tid < 30) {
      int idx2 = S_ + tid - 15;
      s_pad[S_ + tid] = (idx2 < S_) ? prev_attn[b * S_ + idx2] : 0.0f;
    }
  }
  __syncthreads();

  // prenet layer 1 (256x80)
  {
    float acc = pb1[tid];
    const float* wr = pw1 + tid * MEL_;
    for (int k = 0; k < MEL_; ++k) acc += wr[k] * s_dec[k];
    s_pre1[tid] = fmaxf(acc, 0.0f);
  }
  __syncthreads();
  // prenet layer 2 (256x256) -> Acat0[:,0:256]
  {
    float acc = pb2[tid];
    const float* wr = pw2 + tid * PRE_;
    for (int k = 0; k < PRE_; ++k) acc += wr[k] * s_pre1[k];
    Acat0[(size_t)b * 1792 + tid] = f2bf(fmaxf(acc, 0.0f));
  }
  // query q = Wq @ h1  (128x1024)
  if (tid < ATTN_) {
    float acc = 0.0f;
    const float* wr = Wq + tid * DEC_;
    const float* hr = h1f + b * DEC_;
    for (int k = 0; k < DEC_; ++k) acc += wr[k] * hr[k];
    s_q[tid] = acc;
  }
  __syncthreads();

  // location conv (32 ch x 31 taps) per s=tid, then energy
  float loc[32];
#pragma unroll
  for (int c = 0; c < 32; ++c) {
    float acc = loc_b[c];
    const float* wr = loc_w + c * 31;
    for (int k = 0; k < 31; ++k) acc += wr[k] * s_pad[tid + k];
    loc[c] = acc;
  }
  float e = 0.0f;
  {
    const float* kr = keys + (size_t)(b * S_ + tid) * ATTN_;
    for (int a = 0; a < ATTN_; ++a) {
      const float* wl = Wl + a * 32;
      float la = 0.0f;
#pragma unroll
      for (int c = 0; c < 32; ++c) la += wl[c] * loc[c];
      e += vvec[a] * tanhf(kr[a] + s_q[a] + la);
    }
  }
  // softmax over s
  s_red[tid] = e; __syncthreads();
  for (int o = 128; o > 0; o >>= 1) {
    if (tid < o) s_red[tid] = fmaxf(s_red[tid], s_red[tid + o]);
    __syncthreads();
  }
  float mx = s_red[0]; __syncthreads();
  float ex = expf(e - mx);
  s_red[tid] = ex; __syncthreads();
  for (int o = 128; o > 0; o >>= 1) {
    if (tid < o) s_red[tid] += s_red[tid + o];
    __syncthreads();
  }
  float attn = ex / s_red[0];
  s_attn[tid] = attn;
  prev_attn[b * S_ + tid] = attn;
  out_attn[(size_t)(b * T_ + t) * S_ + tid] = attn;
  __syncthreads();

  // context = attn @ enc (512 outputs, 2 per thread); write bf16 to A buffers
  for (int pass = 0; pass < 2; ++pass) {
    int ecol = tid + pass * 256;
    float acc = 0.0f;
    for (int s = 0; s < S_; ++s)
      acc += s_attn[s] * enc[((size_t)(b * S_) + s) * ENC_ + ecol];
    __bf16 cv = f2bf(acc);
    Acat0[(size_t)b * 1792 + PRE_ + ecol] = cv;
    Amel[(size_t)b * (DEC_ + ENC_) + DEC_ + ecol] = cv;
  }
}

// ---------------------------------------------------------------------------
// Postnet conv layer as implicit GEMM with WMMA.
// xT: (B,TP,Cinp) bf16 zero-padded; wT: (Co,5,Cinp) bf16.
// mode 0: y=tanh(g*(conv+cb)*BN_INV+b) -> bf16 xTout (B,TP,Co) at row t+2
// mode 1: final layer -> f32 out_post = conv + cb + mel_residual
// grid: (co_tiles/8, 13 t-tiles, B); 8 waves/block.
// ---------------------------------------------------------------------------
__global__ void __launch_bounds__(256)
k_conv_gemm(const __bf16* __restrict__ xT, const __bf16* __restrict__ wT,
            int Cinp, int Co,
            const float* __restrict__ cb, const float* __restrict__ bn_g,
            const float* __restrict__ bn_b,
            __bf16* __restrict__ xTout,
            float* __restrict__ outFinal, const float* __restrict__ melRes,
            int mode) {
  int lane = threadIdx.x & 31, wave = threadIdx.x >> 5;
  int n0 = (blockIdx.x * 8 + wave) * 16;
  if (n0 >= Co) return;                       // wave-uniform
  int b = blockIdx.z;
  int t0 = blockIdx.y * 16;
  int half = lane >> 4, l15 = lane & 15;
  v8f acc = {};
  const __bf16* wbase = wT + (size_t)(n0 + l15) * (5 * Cinp);
#pragma unroll
  for (int k = 0; k < 5; ++k) {
    const __bf16* arow = xT + (size_t)(b * TP_ + t0 + l15 + k) * Cinp;
    const __bf16* wrow = wbase + k * Cinp;
    for (int c0 = 0; c0 < Cinp; c0 += 32) {
      v16bf a = frag_load(arow, c0, half);
      v16bf w = frag_load(wrow, c0, half);
      acc = __builtin_amdgcn_wmma_f32_16x16x32_bf16(false, a, false, w,
                                                    (short)0, acc, false, false);
    }
  }
  int n = n0 + l15;
#pragma unroll
  for (int r = 0; r < 8; ++r) {
    int t = t0 + r + 8 * half;
    if (t < T_) {
      float y = acc[r] + cb[n];
      if (mode == 0) {
        float hv = tanhf(bn_g[n] * (y * BN_INV_) + bn_b[n]);
        xTout[(size_t)(b * TP_ + t + 2) * Co + n] = f2bf(hv);
      } else {
        size_t oi = (size_t)(b * MEL_ + n) * T_ + t;
        outFinal[oi] = y + melRes[oi];
      }
    }
  }
}

// ---------------------------------------------------------------------------
// Host
// ---------------------------------------------------------------------------
static inline dim3 g1d(long n) { return dim3((unsigned)((n + 255) / 256)); }

extern "C" void kernel_launch(void* const* d_in, const int* in_sizes, int n_in,
                              void* d_out, int out_size, void* d_ws, size_t ws_size,
                              hipStream_t stream) {
  const float* enc   = (const float*)d_in[0];
  const float* melT  = (const float*)d_in[1];
  const float* pw1   = (const float*)d_in[2];
  const float* pb1   = (const float*)d_in[3];
  const float* pw2   = (const float*)d_in[4];
  const float* pb2   = (const float*)d_in[5];
  const float* Wq    = (const float*)d_in[6];
  const float* Wk    = (const float*)d_in[7];
  const float* loc_w = (const float*)d_in[8];
  const float* loc_b = (const float*)d_in[9];
  const float* Wl    = (const float*)d_in[10];
  const float* vvec  = (const float*)d_in[11];
  const float* wih0  = (const float*)d_in[12];
  const float* whh0  = (const float*)d_in[13];
  const float* bih0  = (const float*)d_in[14];
  const float* bhh0  = (const float*)d_in[15];
  const float* wih1  = (const float*)d_in[16];
  const float* whh1  = (const float*)d_in[17];
  const float* bih1  = (const float*)d_in[18];
  const float* bhh1  = (const float*)d_in[19];
  const float* mw    = (const float*)d_in[20];
  const float* mb    = (const float*)d_in[21];
  const float* cw1   = (const float*)d_in[22];
  const float* cb1   = (const float*)d_in[23];
  const float* cw2   = (const float*)d_in[24];
  const float* cb2   = (const float*)d_in[25];
  const float* cw3   = (const float*)d_in[26];
  const float* cb3   = (const float*)d_in[27];
  const float* cw4   = (const float*)d_in[28];
  const float* cb4   = (const float*)d_in[29];
  const float* cw5   = (const float*)d_in[30];
  const float* cb5   = (const float*)d_in[31];
  const float* g1    = (const float*)d_in[32];
  const float* b1    = (const float*)d_in[33];
  const float* g2    = (const float*)d_in[34];
  const float* b2    = (const float*)d_in[35];
  const float* g3    = (const float*)d_in[36];
  const float* b3    = (const float*)d_in[37];
  const float* g4    = (const float*)d_in[38];
  const float* b4    = (const float*)d_in[39];

  float* out_mel  = (float*)d_out;                 // (B,MEL,T)
  float* out_post = out_mel + B_ * MEL_ * T_;      // (B,MEL,T)
  float* out_attn = out_post + B_ * MEL_ * T_;     // (B,T,S)

  char* ws = (char*)d_ws;
  size_t off = 0;
  auto wsalloc = [&](size_t bytes) -> void* {
    off = (off + 255) & ~(size_t)255;
    void* p = ws + off;
    off += bytes;
    return p;
  };
  __bf16* encbf = (__bf16*)wsalloc((size_t)B_ * S_ * ENC_ * 2);
  __bf16* wkbf  = (__bf16*)wsalloc((size_t)ATTN_ * ENC_ * 2);
  float*  keys  = (float*)wsalloc((size_t)B_ * S_ * ATTN_ * 4);
  __bf16* Wcat0 = (__bf16*)wsalloc((size_t)4096 * 1792 * 2);
  __bf16* Wcat1 = (__bf16*)wsalloc((size_t)4096 * 2048 * 2);
  __bf16* mwbf  = (__bf16*)wsalloc((size_t)MEL_ * (DEC_ + ENC_) * 2);
  __bf16* Acat0 = (__bf16*)wsalloc((size_t)B_ * 1792 * 2);
  __bf16* Acat1 = (__bf16*)wsalloc((size_t)B_ * 2048 * 2);
  __bf16* Amel  = (__bf16*)wsalloc((size_t)B_ * (DEC_ + ENC_) * 2);
  float*  z0    = (float*)wsalloc((size_t)B_ * 4096 * 4);
  float*  z1    = (float*)wsalloc((size_t)B_ * 4096 * 4);
  float*  c0    = (float*)wsalloc((size_t)B_ * DEC_ * 4);
  float*  c1    = (float*)wsalloc((size_t)B_ * DEC_ * 4);
  float*  h1f   = (float*)wsalloc((size_t)B_ * DEC_ * 4);
  float*  prevA = (float*)wsalloc((size_t)B_ * S_ * 4);
  __bf16* xTmel = (__bf16*)wsalloc((size_t)B_ * TP_ * CIN1P_ * 2);
  __bf16* xTa   = (__bf16*)wsalloc((size_t)B_ * TP_ * 512 * 2);
  __bf16* xTb   = (__bf16*)wsalloc((size_t)B_ * TP_ * 512 * 2);
  __bf16* wT1   = (__bf16*)wsalloc((size_t)512 * 5 * CIN1P_ * 2);
  __bf16* wT2   = (__bf16*)wsalloc((size_t)512 * 5 * 512 * 2);
  __bf16* wT3   = (__bf16*)wsalloc((size_t)512 * 5 * 512 * 2);
  __bf16* wT4   = (__bf16*)wsalloc((size_t)512 * 5 * 512 * 2);
  __bf16* wT5   = (__bf16*)wsalloc((size_t)MEL_ * 5 * 512 * 2);

  // --- zero state & padded buffers (every call: harness does not re-init) ---
  auto zero32 = [&](void* p, long n32) {
    k_zero<<<g1d(n32), 256, 0, stream>>>((uint32_t*)p, (int)n32);
  };
  zero32(Acat0, (long)B_ * 1792 / 2);
  zero32(Acat1, (long)B_ * 2048 / 2);
  zero32(Amel,  (long)B_ * (DEC_ + ENC_) / 2);
  zero32(c0,    (long)B_ * DEC_);
  zero32(c1,    (long)B_ * DEC_);
  zero32(h1f,   (long)B_ * DEC_);
  zero32(prevA, (long)B_ * S_);
  zero32(xTmel, (long)B_ * TP_ * CIN1P_ / 2);
  zero32(xTa,   (long)B_ * TP_ * 512 / 2);
  zero32(xTb,   (long)B_ * TP_ * 512 / 2);

  // --- weight conversions (bf16, L2-resident for the whole loop) ---
  long nE = (long)B_ * S_ * ENC_;
  k_f2bf<<<g1d(nE), 256, 0, stream>>>(encbf, enc, (int)nE);
  k_f2bf<<<g1d(ATTN_ * ENC_), 256, 0, stream>>>(wkbf, Wk, ATTN_ * ENC_);
  k_f2bf<<<g1d(MEL_ * (DEC_ + ENC_)), 256, 0, stream>>>(mwbf, mw, MEL_ * (DEC_ + ENC_));
  k_concat2<<<g1d((long)4096 * 1792), 256, 0, stream>>>(Wcat0, wih0, 768, whh0, 1024, 4096);
  k_concat2<<<g1d((long)4096 * 2048), 256, 0, stream>>>(Wcat1, wih1, 1024, whh1, 1024, 4096);
  k_convw<<<g1d((long)512 * 5 * CIN1P_), 256, 0, stream>>>(wT1, cw1, 512, MEL_, CIN1P_);
  k_convw<<<g1d((long)512 * 5 * 512), 256, 0, stream>>>(wT2, cw2, 512, 512, 512);
  k_convw<<<g1d((long)512 * 5 * 512), 256, 0, stream>>>(wT3, cw3, 512, 512, 512);
  k_convw<<<g1d((long)512 * 5 * 512), 256, 0, stream>>>(wT4, cw4, 512, 512, 512);
  k_convw<<<g1d((long)MEL_ * 5 * 512), 256, 0, stream>>>(wT5, cw5, MEL_, 512, 512);

  // --- keys = enc @ Wk^T (loop-invariant) ---
  k_keys<<<dim3(B_ * S_ / 16), 256, 0, stream>>>(encbf, wkbf, keys);

  // --- sequential decoder loop ---
  for (int t = 0; t < T_; ++t) {
    k_attention<<<B_, 256, 0, stream>>>(t, melT, enc, pw1, pb1, pw2, pb2, Wq,
                                        keys, loc_w, loc_b, Wl, vvec, h1f,
                                        prevA, out_attn, Acat0, Amel);
    k_gemm16<<<32, 256, 0, stream>>>(Acat0, Wcat0, z0, 1792, 4096);
    k_lstm_pw<<<64, 256, 0, stream>>>(z0, bih0, bhh0, c0, nullptr,
                                      Acat1, 2048, 0, Acat0, 1792, 768);
    k_gemm16<<<32, 256, 0, stream>>>(Acat1, Wcat1, z1, 2048, 4096);
    k_lstm_pw<<<64, 256, 0, stream>>>(z1, bih1, bhh1, c1, h1f,
                                      Amel, DEC_ + ENC_, 0, Acat1, 2048, 1024);
    k_mel<<<1, 256, 0, stream>>>(t, Amel, mwbf, mb, out_mel);
  }

  // --- postnet ---
  k_mel_to_xT<<<g1d((long)B_ * T_ * MEL_), 256, 0, stream>>>(out_mel, xTmel);
  dim3 cg(4, 13, B_);   // 512 output channels
  k_conv_gemm<<<cg, 256, 0, stream>>>(xTmel, wT1, CIN1P_, 512, cb1, g1, b1,
                                      xTa, nullptr, nullptr, 0);
  k_conv_gemm<<<cg, 256, 0, stream>>>(xTa, wT2, 512, 512, cb2, g2, b2,
                                      xTb, nullptr, nullptr, 0);
  k_conv_gemm<<<cg, 256, 0, stream>>>(xTb, wT3, 512, 512, cb3, g3, b3,
                                      xTa, nullptr, nullptr, 0);
  k_conv_gemm<<<cg, 256, 0, stream>>>(xTa, wT4, 512, 512, cb4, g4, b4,
                                      xTb, nullptr, nullptr, 0);
  dim3 cg5(1, 13, B_);  // 80 output channels (5 tiles)
  k_conv_gemm<<<cg5, 256, 0, stream>>>(xTb, wT5, 512, MEL_, cb5, nullptr, nullptr,
                                       nullptr, out_post, out_mel, 1);
  (void)in_sizes; (void)n_in; (void)out_size; (void)ws_size;
}
